// CAB_85736137163309
// MI455X (gfx1250) — matrix operand order repaired and checked by reading, not compiled, not run on hardware
//
#include <hip/hip_runtime.h>
#include <math.h>

// Problem constants (match reference)
#define M_DIM 262144
#define C_DIM 256

// k1 config
#define K1_BLOCKS 256
#define K1_THREADS 512              // 16 waves
#define K1_ROWS (M_DIM / K1_BLOCKS) // 1024 rows of A per block
#define KCH 32                      // K-chunk (WMMA bf16 K)
#define LDT 36                      // bf16 row stride for 32-wide chunks (72B, 8B aligned, conflict-spread)

// k3 config
#define K3_THREADS 256              // 8 waves
#define K3_ROWS 128                 // rows of A/E per block

typedef __attribute__((ext_vector_type(16))) __bf16 v16bf;
typedef __attribute__((ext_vector_type(8)))  float  v8f;
typedef __attribute__((ext_vector_type(4)))  unsigned int v4u;
typedef __attribute__((ext_vector_type(4)))  int    v4i;
typedef __attribute__((ext_vector_type(8)))  int    v8i;

union FragBF { v16bf v; int2 d[4]; };

__device__ __forceinline__ unsigned short f2bf(float f) {
    unsigned int u = __float_as_uint(f);
    u += 0x7fffu + ((u >> 16) & 1u);   // round-to-nearest-even
    return (unsigned short)(u >> 16);
}

__device__ __forceinline__ unsigned lds_off(const void* p) {
    // Generic (flat) LDS pointers carry the LDS byte offset in the low 32 bits.
    return (unsigned)(unsigned long long)p;
}

// Per-lane async copy: 16B global -> LDS, tracked by ASYNCcnt.
__device__ __forceinline__ void async_load_b128(unsigned ldsAddr, unsigned long long gaddr) {
    asm volatile("global_load_async_to_lds_b128 %0, %1, off"
                 :: "v"(ldsAddr), "v"(gaddr) : "memory");
}
__device__ __forceinline__ void wait_asynccnt0() {
    asm volatile("s_wait_asynccnt 0x0" ::: "memory");
}

// ---------------------------------------------------------------------------
// Generic 2D Tensor-Data-Mover load: tileD1 rows of tileD0 elements,
// row stride tensorStride0 elements, optional LDS destination padding.
// ---------------------------------------------------------------------------
__device__ __forceinline__ void tdm_load_2d(unsigned ldsAddr, const void* gsrc,
                                            unsigned dsCode,        // 0..3 => 1,2,4,8 B
                                            unsigned tensorD0, unsigned tensorD1,
                                            unsigned tileD0, unsigned tileD1,
                                            unsigned tensorStride0,
                                            unsigned padEn,
                                            unsigned padIntCode, unsigned padAmtCode) {
    unsigned long long ga = (unsigned long long)gsrc;
    v4u g0;
    g0.x = 1u;                                   // count=1, user descriptor
    g0.y = ldsAddr;                              // lds_addr (bytes)
    g0.z = (unsigned)(ga & 0xffffffffu);         // global_addr[31:0]
    g0.w = (unsigned)((ga >> 32) & 0x01ffffffu)  // global_addr[56:32]
         | (2u << 30);                           // type=2 ("image")
    v8i g1;
    g1[0] = (int)((dsCode << 16) | (padEn << 20) |
                  (padIntCode << 22) | (padAmtCode << 25));
    g1[1] = (int)((tensorD0 & 0xffffu) << 16);                    // tensor_dim0 lo (bits 79:48)
    g1[2] = (int)((tensorD0 >> 16) | ((tensorD1 & 0xffffu) << 16)); // dim0 hi | dim1 lo
    g1[3] = (int)((tensorD1 >> 16) | (tileD0 << 16));             // dim1 hi | tile_dim0
    g1[4] = (int)tileD1;                                          // tile_dim1 | tile_dim2=0
    g1[5] = (int)tensorStride0;                                   // dim0 stride lo
    g1[6] = 0;                                                    // stride hi | dim1 stride lo
    g1[7] = 0;
    v4i gz = {0, 0, 0, 0};
#if __clang_major__ >= 23
    v8i gz8 = {0, 0, 0, 0, 0, 0, 0, 0};
    __builtin_amdgcn_tensor_load_to_lds(g0, g1, gz, gz, gz8, 0);
#else
    __builtin_amdgcn_tensor_load_to_lds(g0, g1, gz, gz, 0);
#endif
}

// ---------------------------------------------------------------------------
// k1: partial[b] = (A_chunk)^T * (A_chunk)  for this block's 1024-row chunk.
// Each of 16 waves owns a 16x256 strip of the 256x256 output.
// K-chunks (32x256 f32, 32KB contiguous) are DMA'd by the TDM into a raw LDS
// buffer (double-clocked against compute), then transposed/converted to bf16.
// ---------------------------------------------------------------------------
__global__ __launch_bounds__(K1_THREADS)
void k1_syrk(const float* __restrict__ A, float* __restrict__ partial) {
    __shared__ float          raw [KCH * C_DIM];   // TDM dest, 32768 B
    __shared__ unsigned short ldsT[C_DIM * LDT];   // A-chunk transposed bf16 [c][k], 18432 B

    const int tid  = threadIdx.x;
    const int lane = tid & 31;
    const int wave = tid >> 5;      // 0..15
    const int r    = lane & 15;
    const int h    = lane >> 4;     // lane half
    const int c0   = wave * 16;     // this wave's output-row strip

    v8f acc[16] = {};

    const size_t rowBase = (size_t)blockIdx.x * K1_ROWS;
    const unsigned rawAddr = lds_off(&raw[0]);

    // Prologue: DMA first chunk (one 1 x 8192-elem f32 tile, fully contiguous).
    if (wave == 0)
        tdm_load_2d(rawAddr, A + rowBase * C_DIM, 2u,
                    KCH * C_DIM, 1u, KCH * C_DIM, 1u, KCH * C_DIM, 0u, 0u, 0u);

    for (int kc = 0; kc < K1_ROWS; kc += KCH) {
        const bool hasNext = (kc + KCH) < K1_ROWS;
        if (wave == 0) __builtin_amdgcn_s_wait_tensorcnt(0);  // chunk kc landed
        __syncthreads();            // raw visible to all; prev compute done

        // Transpose/convert raw f32 [k][c] -> bf16 ldsT [c][k]
#pragma unroll
        for (int i = 0; i < (KCH * C_DIM) / K1_THREADS; ++i) {
            int idx = tid + i * K1_THREADS;   // == k*256 + c
            int k = idx >> 8;
            int c = idx & 255;
            ldsT[c * LDT + k] = f2bf(raw[idx]);
        }
        __syncthreads();            // ldsT ready; raw free for next DMA

        // Stream next chunk while this one is consumed by the WMMAs.
        if (hasNext && wave == 0)
            tdm_load_2d(rawAddr, A + (rowBase + kc + KCH) * C_DIM, 2u,
                        KCH * C_DIM, 1u, KCH * C_DIM, 1u, KCH * C_DIM, 0u, 0u, 0u);

        // A fragment (16x32): lane r -> row c0+r of A^T; K run = [16h, 16h+16)
        FragBF fa;
        {
            const int2* row = (const int2*)(ldsT + (c0 + r) * LDT);
            fa.d[0] = row[4 * h + 0];
            fa.d[1] = row[4 * h + 1];
            fa.d[2] = row[4 * h + 2];
            fa.d[3] = row[4 * h + 3];
        }
#pragma unroll
        for (int t = 0; t < 16; ++t) {
            // B fragment (32x16): lane r -> column 16t+r; K run = [16h, 16h+16)
            FragBF fb;
            const int2* row = (const int2*)(ldsT + (t * 16 + r) * LDT);
            fb.d[0] = row[4 * h + 0];
            fb.d[1] = row[4 * h + 1];
            fb.d[2] = row[4 * h + 2];
            fb.d[3] = row[4 * h + 3];
            acc[t] = __builtin_amdgcn_wmma_f32_16x16x32_bf16(
                false, fa.v, false, fb.v, (short)0, acc[t], false, false);
        }
    }

    // Store this block's 256x256 f32 partial
    float* base = partial + (size_t)blockIdx.x * (C_DIM * C_DIM);
#pragma unroll
    for (int t = 0; t < 16; ++t) {
#pragma unroll
        for (int j = 0; j < 8; ++j) {
            base[(c0 + j + 8 * h) * C_DIM + (t * 16 + r)] = acc[t][j];
        }
    }
}

// ---------------------------------------------------------------------------
// k2: interaction row = sum of partials; row softmax -> CAG (f32) + bf16 copy
// ---------------------------------------------------------------------------
__global__ __launch_bounds__(256)
void k2_reduce_softmax(const float* __restrict__ partial, float* __restrict__ cag,
                       unsigned short* __restrict__ cagBf) {
    __shared__ float red[256];
    const int d   = threadIdx.x;
    const int row = blockIdx.x;

    float v = 0.0f;
    const float* p = partial + (size_t)row * C_DIM + d;
    for (int b = 0; b < K1_BLOCKS; ++b)
        v += p[(size_t)b * (C_DIM * C_DIM)];

    red[d] = v;
    __syncthreads();
    for (int s = 128; s > 0; s >>= 1) {
        if (d < s) red[d] = fmaxf(red[d], red[d + s]);
        __syncthreads();
    }
    float mx = red[0];
    __syncthreads();

    float e = __expf(v - mx);
    red[d] = e;
    __syncthreads();
    for (int s = 128; s > 0; s >>= 1) {
        if (d < s) red[d] += red[d + s];
        __syncthreads();
    }
    float out = e / red[0];
    cag[(size_t)row * C_DIM + d]   = out;
    cagBf[(size_t)row * C_DIM + d] = f2bf(out);
}

// ---------------------------------------------------------------------------
// k3: E = A + beta * (A @ CAG^T). 128 rows/block; 8 waves x 16-row strips.
// CAG slices arrive via TDM (double-buffered, padded rows); A tiles stream in
// via per-lane async global->LDS copies overlapped with the WMMAs.
// ---------------------------------------------------------------------------
__global__ __launch_bounds__(K3_THREADS)
void k3_egemm(const float* __restrict__ A, const unsigned short* __restrict__ cagBf,
              const float* __restrict__ betaPtr, float* __restrict__ E) {
    __shared__ unsigned short lds_cg[2][C_DIM * LDT];  // CAG k-slices (TDM dest) 2x18432B
    __shared__ unsigned short lds_a[K3_ROWS * LDT];    // A tile bf16              9216B
    __shared__ float          raw_a[K3_ROWS * KCH];    // A tile raw f32          16384B

    const int tid  = threadIdx.x;
    const int lane = tid & 31;
    const int wave = tid >> 5;      // 0..7
    const int r    = lane & 15;
    const int h    = lane >> 4;
    const size_t mBase = (size_t)blockIdx.x * K3_ROWS;

    const unsigned rawBase = lds_off(&raw_a[0]);

    // Issue this wave's share (rows [16*wave, 16*wave+16)) of the raw A tile.
    auto issue_a_tile = [&](int k0) {
#pragma unroll
        for (int j = 0; j < 4; ++j) {
            int linear = j * 32 + lane;        // 0..127
            int rowL   = linear >> 3;          // 0..15
            int p      = linear & 7;           // 16B chunk (row = 32 f32 = 128B)
            int row    = wave * 16 + rowL;
            unsigned long long g =
                (unsigned long long)(const void*)(A + (mBase + row) * C_DIM + k0) +
                (unsigned long long)(p * 16);
            async_load_b128(rawBase + (unsigned)(row * 128 + p * 16), g);
        }
    };

    // Prologue: kick off first A tile + first CAG slice.
    issue_a_tile(0);
    if (wave == 0)
        tdm_load_2d(lds_off(&lds_cg[0][0]), cagBf, 1u,
                    C_DIM, C_DIM, KCH, C_DIM, C_DIM, 1u, 3u, 1u);

    v8f acc[16] = {};
    int cur = 0;

    for (int k0 = 0; k0 < C_DIM; k0 += KCH) {
        const bool hasNext = (k0 + KCH) < C_DIM;
        __syncthreads();            // previous compute done: lds_a & lds_cg[!cur] free
        wait_asynccnt0();           // this wave's raw A region for k0 has landed

        // Convert my region raw f32 -> bf16 lds_a
#pragma unroll
        for (int i = 0; i < 16; ++i) {
            int e    = i * 32 + lane;          // 0..511
            int rowL = e >> 5;
            int k    = e & 31;
            int row  = wave * 16 + rowL;
            lds_a[row * LDT + k] = f2bf(raw_a[row * KCH + k]);
        }
        // My raw region is consumed: start streaming the next tile into it.
        if (hasNext) issue_a_tile(k0 + KCH);
        if (wave == 0) {
            if (hasNext) {
                tdm_load_2d(lds_off(&lds_cg[cur ^ 1][0]), cagBf + (k0 + KCH), 1u,
                            C_DIM, C_DIM, KCH, C_DIM, C_DIM, 1u, 3u, 1u);
                __builtin_amdgcn_s_wait_tensorcnt(1);   // current slice done
            } else {
                __builtin_amdgcn_s_wait_tensorcnt(0);
            }
        }
        __syncthreads();            // lds_a staged + lds_cg[cur] complete (via wave0)

        // A fragment: lane r -> row wave*16+r of the tile
        FragBF fa;
        {
            const int2* row = (const int2*)(lds_a + (wave * 16 + r) * LDT);
            fa.d[0] = row[4 * h + 0];
            fa.d[1] = row[4 * h + 1];
            fa.d[2] = row[4 * h + 2];
            fa.d[3] = row[4 * h + 3];
        }
#pragma unroll
        for (int t = 0; t < 16; ++t) {
            // B[k][n] = CAG[16t+n][k0+k]; lane r -> n=r, contiguous K run [16h,16h+16)
            FragBF fb;
            const int2* row = (const int2*)(&lds_cg[cur][(t * 16 + r) * LDT]);
            fb.d[0] = row[4 * h + 0];
            fb.d[1] = row[4 * h + 1];
            fb.d[2] = row[4 * h + 2];
            fb.d[3] = row[4 * h + 3];
            acc[t] = __builtin_amdgcn_wmma_f32_16x16x32_bf16(
                false, fa.v, false, fb.v, (short)0, acc[t], false, false);
        }
        cur ^= 1;
    }

    const float beta = *betaPtr;
#pragma unroll
    for (int t = 0; t < 16; ++t) {
#pragma unroll
        for (int j = 0; j < 8; ++j) {
            size_t row = mBase + wave * 16 + j + 8 * h;
            size_t idx = row * C_DIM + (t * 16 + r);
            E[idx] = A[idx] + beta * acc[t][j];   // exact f32 epilogue
        }
    }
}

// ---------------------------------------------------------------------------
extern "C" void kernel_launch(void* const* d_in, const int* in_sizes, int n_in,
                              void* d_out, int out_size, void* d_ws, size_t ws_size,
                              hipStream_t stream) {
    (void)in_sizes; (void)n_in; (void)out_size; (void)ws_size;

    const float* A    = (const float*)d_in[0];
    const float* beta = (const float*)d_in[1];

    float* E   = (float*)d_out;                            // M*C floats
    float* cag = (float*)d_out + (size_t)M_DIM * C_DIM;    // C*C floats

    // Split-K partial scratch reuses the not-yet-written E region (64 MB < 256 MB).
    float* partial = E;
    // bf16 CAG copy (128 KB) lives in workspace.
    unsigned short* cagBf = (unsigned short*)d_ws;

    k1_syrk<<<K1_BLOCKS, K1_THREADS, 0, stream>>>(A, partial);
    k2_reduce_softmax<<<C_DIM, C_DIM, 0, stream>>>(partial, cag, cagBf);
    k3_egemm<<<M_DIM / K3_ROWS, K3_THREADS, 0, stream>>>(A, cagBf, beta, E);
}
